// DeepAttention_57114475102770
// MI455X (gfx1250) — matrix operand order repaired and checked by reading, not compiled, not run on hardware
//
#include <hip/hip_runtime.h>
#include <hip/hip_bf16.h>

// ---------------------------------------------------------------------------
// Types / WMMA helpers (CDNA5 gfx1250, wave32)
// ---------------------------------------------------------------------------
typedef __attribute__((ext_vector_type(16))) __bf16 v16bf;
typedef __attribute__((ext_vector_type(8)))  float  v8f;

union Frag { v16bf v; unsigned int u[8]; };

__device__ __forceinline__ unsigned short f2bf(float x) {
  unsigned int u = __float_as_uint(x);
  return (unsigned short)((u + 0x7FFFu + ((u >> 16) & 1u)) >> 16);  // RNE
}

// A fragment: 16x32 bf16 (MxK). Lane holds row m = lane&15.
// lanes 0-15: VGPR0..7 -> K 0..15; lanes 16-31: K 16..31.
__device__ __forceinline__ v16bf load_a(const unsigned short* rowk) {
  const int khalf = ((threadIdx.x & 31) >> 4) * 16;
  Frag f;
#pragma unroll
  for (int v = 0; v < 8; ++v) {
    const int k = 2 * v + khalf;
    f.u[v] = *(const unsigned int*)(rowk + k);
  }
  return f.v;
}

// B fragment from weight-style storage W[n][k] (row-major [N][K]).
__device__ __forceinline__ v16bf load_b_nmajor(const unsigned short* W, size_t ldb,
                                               int n0, int k0) {
  const int lane = threadIdx.x & 31;
  const unsigned short* p = W + (size_t)(n0 + (lane & 15)) * ldb + k0 + ((lane >> 4) << 4);
  Frag f;
#pragma unroll
  for (int v = 0; v < 8; ++v) f.u[v] = *(const unsigned int*)(p + 2 * v);
  return f.v;
}

// B fragment from activation-style storage X[k][n] (row-major [K][N]).
__device__ __forceinline__ v16bf load_b_kmajor(const unsigned short* X, size_t ldb,
                                               int n0, int k0) {
  const int lane = threadIdx.x & 31;
  const unsigned short* p = X + (size_t)(k0 + ((lane >> 4) << 4)) * ldb + n0 + (lane & 15);
  Frag f;
#pragma unroll
  for (int v = 0; v < 8; ++v) {
    const unsigned int lo = p[(size_t)(2 * v) * ldb];
    const unsigned int hi = p[(size_t)(2 * v + 1) * ldb];
    f.u[v] = lo | (hi << 16);
  }
  return f.v;
}

__device__ __forceinline__ v8f wmma_bf16(v16bf a, v16bf b, v8f c) {
  return __builtin_amdgcn_wmma_f32_16x16x32_bf16(false, a, false, b, (short)0, c,
                                                 false, false);
}

// Async global->LDS b128 copy (gfx1250 Tensor/async path) + wait.
__device__ __forceinline__ void async_g2l_b128(unsigned lds_off, const void* gptr) {
  asm volatile("global_load_async_to_lds_b128 %0, %1, off"
               :: "v"(lds_off), "v"(gptr) : "memory");
}

// ---------------------------------------------------------------------------
// Problem constants
// ---------------------------------------------------------------------------
#define BB   32
#define LL   512
#define WORD 300
#define ABH  512
#define ATT  1324
#define ATTP 1344      // 42 * 32
#define DD   250
#define DP   256
#define RIN  2560
#define G4   1024      // 4*H
#define HH   256
#define MROW (BB * LL) // 16384
#define ZKT  (RIN / 32) // 80

// ---------------------------------------------------------------------------
// Elementwise prep kernels
// ---------------------------------------------------------------------------
__global__ void k_f2bf(unsigned short* dst, const float* src, size_t n) {
  size_t i = (size_t)blockIdx.x * blockDim.x + threadIdx.x;
  const size_t st = (size_t)gridDim.x * blockDim.x;
  for (; i < n; i += st) dst[i] = f2bf(src[i]);
}

__global__ void k_pack_att(unsigned short* dst, const float* w, const float* a0,
                           const float* a1) {
  const size_t n = (size_t)MROW * ATTP;
  size_t i = (size_t)blockIdx.x * blockDim.x + threadIdx.x;
  const size_t st = (size_t)gridDim.x * blockDim.x;
  for (; i < n; i += st) {
    const size_t row = i / ATTP;
    const int c = (int)(i % ATTP);
    float v;
    if (c < WORD)            v = w[row * WORD + c];
    else if (c < WORD + ABH) v = a0[row * ABH + (c - WORD)];
    else if (c < ATT)        v = a1[row * ABH + (c - WORD - ABH)];
    else                     v = 0.f;
    dst[i] = f2bf(v);
  }
}

__global__ void k_pack_U(unsigned short* dst, const float* U) {
  const size_t n = (size_t)DP * ATTP;
  size_t i = (size_t)blockIdx.x * blockDim.x + threadIdx.x;
  const size_t st = (size_t)gridDim.x * blockDim.x;
  for (; i < n; i += st) {
    const int d = (int)(i / ATTP), a = (int)(i % ATTP);
    dst[i] = (d < DD && a < ATT) ? f2bf(U[(size_t)d * ATT + a]) : (unsigned short)0;
  }
}

__global__ void k_pack_x1cat(unsigned short* dst, const float* a0, const float* a1) {
  const size_t n = (size_t)MROW * 1024;
  size_t i = (size_t)blockIdx.x * blockDim.x + threadIdx.x;
  const size_t st = (size_t)gridDim.x * blockDim.x;
  for (; i < n; i += st) {
    const size_t row = i / 1024;
    const int c = (int)(i % 1024);
    const float v = (c < ABH) ? a0[row * ABH + c] : a1[row * ABH + (c - ABH)];
    dst[row * RIN + c] = f2bf(v);
  }
}

__global__ void k_bias(float* dst, const float* bih, const float* bhh) {
  const int i = blockIdx.x * blockDim.x + threadIdx.x;
  if (i < G4) dst[i] = bih[i] + bhh[i];
}

// ---------------------------------------------------------------------------
// GEMM 1: r = ReLU(x_att @ U^T) [* d]   M=16384, N=256, K=1344
// One wave -> 16x64 output slab (A fragment reused across 4 N-tiles).
// ---------------------------------------------------------------------------
__global__ __launch_bounds__(32) void k_proj(const unsigned short* __restrict__ Xa,
                       const unsigned short* __restrict__ Ub,
                       const float* __restrict__ dvec,
                       unsigned short* __restrict__ Rout, int scaled) {
  const int nt = blockIdx.x, mt = blockIdx.y;   // nt: groups of 64 cols
  const int lane = threadIdx.x & 31;
  const unsigned short* arow = Xa + (size_t)(mt * 16 + (lane & 15)) * ATTP;
  v8f acc[4] = {{}, {}, {}, {}};
#pragma unroll 1
  for (int kk = 0; kk < ATTP / 32; ++kk) {
    const v16bf a = load_a(arow + kk * 32);
#pragma unroll
    for (int tn = 0; tn < 4; ++tn)
      acc[tn] = wmma_bf16(a, load_b_nmajor(Ub, ATTP, nt * 64 + tn * 16, kk * 32), acc[tn]);
  }
  const int m0 = mt * 16 + ((lane >> 4) << 3);
#pragma unroll
  for (int tn = 0; tn < 4; ++tn) {
    const int n = nt * 64 + tn * 16 + (lane & 15);
    const float sc = scaled ? ((n < DD) ? dvec[n] : 0.f) : 1.f;
#pragma unroll
    for (int r = 0; r < 8; ++r) {
      float v = acc[tn][r];
      v = (v > 0.f) ? v : 0.f;
      Rout[(size_t)(m0 + r) * DP + n] = f2bf(v * sc);
    }
  }
}

// ---------------------------------------------------------------------------
// GEMM 2: scores[l,m] = r1 @ r2^T  (K=256), masked -> f32. 16x64 per wave.
// ---------------------------------------------------------------------------
__global__ __launch_bounds__(32) void k_scores(const unsigned short* __restrict__ R1,
                         const unsigned short* __restrict__ R2,
                         const unsigned char* __restrict__ mask,
                         float* __restrict__ S) {
  const int mt = blockIdx.x, lt = blockIdx.y, hb = blockIdx.z;  // hb = head*32 + b
  const int b = hb & 31;
  const int lane = threadIdx.x & 31;
  const unsigned short* Abase = R1 + (size_t)hb * LL * DP;
  const unsigned short* Bbase = R2 + (size_t)hb * LL * DP;
  const unsigned short* arow = Abase + (size_t)(lt * 16 + (lane & 15)) * DP;
  v8f acc[4] = {{}, {}, {}, {}};
#pragma unroll 1
  for (int kk = 0; kk < DP / 32; ++kk) {
    const v16bf a = load_a(arow + kk * 32);
#pragma unroll
    for (int tn = 0; tn < 4; ++tn)
      acc[tn] = wmma_bf16(a, load_b_nmajor(Bbase, DP, mt * 64 + tn * 16, kk * 32), acc[tn]);
  }
  const int l0 = lt * 16 + ((lane >> 4) << 3);
  float* Sr = S + (size_t)hb * LL * LL;
#pragma unroll
  for (int tn = 0; tn < 4; ++tn) {
    const int m = mt * 64 + tn * 16 + (lane & 15);
    const int masked = mask[(size_t)b * LL + m];
#pragma unroll
    for (int r = 0; r < 8; ++r)
      Sr[(size_t)(l0 + r) * LL + m] = masked ? -__builtin_inff() : acc[tn][r];
  }
}

// ---------------------------------------------------------------------------
// Softmax: one wave per row of 512; write bf16 alpha in-place (row stride 1024 bf16)
// ---------------------------------------------------------------------------
__global__ void k_softmax(float* S) {
  const size_t row = (size_t)blockIdx.x * (blockDim.x >> 5) + (threadIdx.x >> 5);
  const int lane = threadIdx.x & 31;
  float* rp = S + row * LL;
  float v[16];
  float mx = -__builtin_inff();
#pragma unroll
  for (int i = 0; i < 16; ++i) { v[i] = rp[lane + 32 * i]; mx = fmaxf(mx, v[i]); }
#pragma unroll
  for (int o = 16; o; o >>= 1) mx = fmaxf(mx, __shfl_xor(mx, o, 32));
  float sum = 0.f;
#pragma unroll
  for (int i = 0; i < 16; ++i) { v[i] = __expf(v[i] - mx); sum += v[i]; }
#pragma unroll
  for (int o = 16; o; o >>= 1) sum += __shfl_xor(sum, o, 32);
  const float inv = 1.f / sum;
  unsigned short* pw = (unsigned short*)rp;  // first half of the 2048B f32 row
#pragma unroll
  for (int i = 0; i < 16; ++i) pw[lane + 32 * i] = f2bf(v[i] * inv);
}

// ---------------------------------------------------------------------------
// GEMM 3: out = alpha @ x3  (K=512) -> bf16 into x1cat slice. 16x64 per wave.
// ---------------------------------------------------------------------------
__global__ __launch_bounds__(32) void k_attnv(const unsigned short* __restrict__ P,  // rows ld=1024
                        const unsigned short* __restrict__ X3,
                        unsigned short* __restrict__ X1c, int head) {
  const int ht = blockIdx.x, lt = blockIdx.y, b = blockIdx.z;
  const int lane = threadIdx.x & 31;
  const unsigned short* Abase = P + ((size_t)head * BB + b) * LL * 1024;
  const unsigned short* arow = Abase + (size_t)(lt * 16 + (lane & 15)) * 1024;
  const unsigned short* Bbase = X3 + ((size_t)head * BB + b) * LL * ABH;  // [m][h]
  v8f acc[4] = {{}, {}, {}, {}};
#pragma unroll 1
  for (int kk = 0; kk < LL / 32; ++kk) {
    const v16bf a = load_a(arow + kk * 32);
#pragma unroll
    for (int tn = 0; tn < 4; ++tn)
      acc[tn] = wmma_bf16(a, load_b_kmajor(Bbase, ABH, ht * 64 + tn * 16, kk * 32), acc[tn]);
  }
  const int l0 = lt * 16 + ((lane >> 4) << 3);
  unsigned short* out = X1c + (size_t)b * LL * RIN;
#pragma unroll
  for (int tn = 0; tn < 4; ++tn) {
    const int n = ht * 64 + tn * 16 + (lane & 15);
#pragma unroll
    for (int r = 0; r < 8; ++r)
      out[(size_t)(l0 + r) * RIN + 1024 + head * ABH + n] = f2bf(acc[tn][r]);
  }
}

// ---------------------------------------------------------------------------
// GEMM 4: Z[dir][t][b][:] = x1cat[b,t,:] @ Wih^T + (bih+bhh)   K=2560
// Async double-buffered LDS staging: GLOBAL_LOAD_ASYNC_TO_LDS_B128 fills
// tile-set k+1 while WMMA consumes tile-set k from LDS. 10 b128 copies per set
// (2 for the 16x32 A tile, 8 for four 16x32 B tiles); ASYNCcnt completes in
// order, so s_wait_asynccnt 10 guarantees the current set is resident.
// ---------------------------------------------------------------------------
__global__ __launch_bounds__(32) void k_zproj(const unsigned short* __restrict__ X1c,
                        const unsigned short* __restrict__ Wihb,
                        const float* __restrict__ bias, float* __restrict__ Z) {
  const int nt = blockIdx.x;      // 0..15: groups of 64 cols
  const int my = blockIdx.y;      // 0..1023
  const int dir = blockIdx.z;
  const int lane = threadIdx.x & 31;
  const int t = my >> 1;
  const int b0 = (my & 1) * 16;
  const unsigned short* W = Wihb + (size_t)dir * G4 * RIN;

  __shared__ unsigned short tA[2][16][32];
  __shared__ unsigned short tB[2][4][16][32];

  const int rr = lane & 15;
  const int c0 = (lane >> 4) * 2;  // this lane stages 16B chunks c0, c0+1 of a 64B row
  const unsigned short* gA = X1c + ((size_t)(b0 + rr) * LL + t) * RIN;

  auto stage = [&](int buf, int kk) {
    const unsigned short* ga = gA + kk * 32 + c0 * 8;
    const unsigned la = (unsigned)(uintptr_t)&tA[buf][rr][c0 * 8];
    async_g2l_b128(la, ga);
    async_g2l_b128(la + 16u, ga + 8);
#pragma unroll
    for (int tn = 0; tn < 4; ++tn) {
      const unsigned short* gb = W + (size_t)(nt * 64 + tn * 16 + rr) * RIN + kk * 32 + c0 * 8;
      const unsigned lb = (unsigned)(uintptr_t)&tB[buf][tn][rr][c0 * 8];
      async_g2l_b128(lb, gb);
      async_g2l_b128(lb + 16u, gb + 8);
    }
  };

  v8f acc[4] = {{}, {}, {}, {}};
  stage(0, 0);
#pragma unroll 1
  for (int kk = 0; kk < ZKT; ++kk) {
    const int cur = kk & 1;
    if (kk + 1 < ZKT) {
      stage(cur ^ 1, kk + 1);
      asm volatile("s_wait_asynccnt 0xa" ::: "memory");  // current set resident
    } else {
      asm volatile("s_wait_asynccnt 0x0" ::: "memory");
    }
    const v16bf a = load_a(&tA[cur][lane & 15][0]);
#pragma unroll
    for (int tn = 0; tn < 4; ++tn)
      acc[tn] = wmma_bf16(a, load_b_nmajor(&tB[cur][tn][0][0], 32, 0, 0), acc[tn]);
  }

  const int m0 = b0 + ((lane >> 4) << 3);
  float* zrow = Z + (((size_t)dir * LL + t) * BB) * G4;
#pragma unroll
  for (int tn = 0; tn < 4; ++tn) {
    const int n = nt * 64 + tn * 16 + (lane & 15);
    const float bs = bias[dir * G4 + n];
#pragma unroll
    for (int r = 0; r < 8; ++r) zrow[(size_t)(m0 + r) * G4 + n] = acc[tn][r] + bs;
  }
}

// ---------------------------------------------------------------------------
// LSTM recurrence: persistent block per direction. h in LDS (bf16), c in regs.
// 32 waves: wave = (jt<<1)|bhalf; each wave owns i/f/g/o tiles for its (b,j) slab.
// ---------------------------------------------------------------------------
__global__ __launch_bounds__(1024) void k_lstm(const float* __restrict__ Z,
                                               const unsigned short* __restrict__ Whhb,
                                               float* __restrict__ out) {
  const int dir = blockIdx.x;
  const int tid = threadIdx.x;
  const int lane = tid & 31;
  const int wave = tid >> 5;
  const int jt = wave >> 1;
  const int b0 = (wave & 1) * 16;
  __shared__ unsigned short hbuf[BB][HH + 8];  // padded, 4B-aligned rows
  for (int i = tid; i < BB * (HH + 8); i += 1024) ((unsigned short*)hbuf)[i] = 0;
  __syncthreads();
  const unsigned short* W = Whhb + (size_t)dir * G4 * HH;
  const int m0 = b0 + ((lane >> 4) << 3);
  const int j = jt * 16 + (lane & 15);
  float c[8];
#pragma unroll
  for (int r = 0; r < 8; ++r) c[r] = 0.f;

  for (int s = 0; s < LL; ++s) {
    const int t = dir ? (LL - 1 - s) : s;
    v8f aI = {}, aF = {}, aG = {}, aO = {};
#pragma unroll 1
    for (int kk = 0; kk < HH / 32; ++kk) {
      const v16bf afr = load_a(&hbuf[b0 + (lane & 15)][kk * 32]);
      aI = wmma_bf16(afr, load_b_nmajor(W, HH, 0 * HH + jt * 16, kk * 32), aI);
      aF = wmma_bf16(afr, load_b_nmajor(W, HH, 1 * HH + jt * 16, kk * 32), aF);
      aG = wmma_bf16(afr, load_b_nmajor(W, HH, 2 * HH + jt * 16, kk * 32), aG);
      aO = wmma_bf16(afr, load_b_nmajor(W, HH, 3 * HH + jt * 16, kk * 32), aO);
    }
    __syncthreads();  // all hbuf reads complete before updates
    const float* zb = Z + (((size_t)dir * LL + t) * BB) * G4;
#pragma unroll
    for (int r = 0; r < 8; ++r) {
      const int bidx = m0 + r;
      const float* zr = zb + (size_t)bidx * G4;
      const float zi = zr[0 * HH + j] + aI[r];
      const float zf = zr[1 * HH + j] + aF[r];
      const float zg = zr[2 * HH + j] + aG[r];
      const float zo = zr[3 * HH + j] + aO[r];
      const float ig = 1.f / (1.f + __expf(-zi));
      const float fg = 1.f / (1.f + __expf(-zf));
      const float og = 1.f / (1.f + __expf(-zo));
      c[r] = fg * c[r] + ig * tanhf(zg);
      const float hv = og * tanhf(c[r]);
      hbuf[bidx][j] = f2bf(hv);
      out[((size_t)bidx * LL + t) * (2 * HH) + dir * HH + j] = hv;
    }
    __syncthreads();  // updates visible before next step's fragment loads
  }
}

// ---------------------------------------------------------------------------
// Launcher
// ---------------------------------------------------------------------------
static inline size_t alignup(size_t x) { return (x + 255) & ~(size_t)255; }

extern "C" void kernel_launch(void* const* d_in, const int* in_sizes, int n_in,
                              void* d_out, int out_size, void* d_ws, size_t ws_size,
                              hipStream_t stream) {
  (void)in_sizes; (void)n_in; (void)out_size; (void)ws_size;
  const float* x1w  = (const float*)d_in[0];
  const float* x1a0 = (const float*)d_in[1];
  const float* x1a1 = (const float*)d_in[2];
  const float* x2w  = (const float*)d_in[3];
  const float* x2a0 = (const float*)d_in[4];
  const float* x2a1 = (const float*)d_in[5];
  const float* x2a2 = (const float*)d_in[6];
  const unsigned char* mask = (const unsigned char*)d_in[8];
  const float* Us[3] = {(const float*)d_in[9], (const float*)d_in[11], (const float*)d_in[13]};
  const float* ds[3] = {(const float*)d_in[10], (const float*)d_in[12], (const float*)d_in[14]};
  const float* Wih_f = (const float*)d_in[15];
  const float* Whh_f = (const float*)d_in[16];
  const float* bih_f = (const float*)d_in[17];
  const float* bhh_f = (const float*)d_in[18];
  const float* Wih_b = (const float*)d_in[19];
  const float* Whh_b = (const float*)d_in[20];
  const float* bih_b = (const float*)d_in[21];
  const float* bhh_b = (const float*)d_in[22];

  char* w = (char*)d_ws;
  size_t off = 0;
  auto take = [&](size_t bytes) -> void* { void* p = w + off; off = alignup(off + bytes); return p; };
  unsigned short* x1a = (unsigned short*)take((size_t)MROW * ATTP * 2);
  unsigned short* x2a = (unsigned short*)take((size_t)MROW * ATTP * 2);
  unsigned short* Ubf = (unsigned short*)take((size_t)3 * DP * ATTP * 2);
  unsigned short* R1  = (unsigned short*)take((size_t)3 * MROW * DP * 2);
  unsigned short* R2  = (unsigned short*)take((size_t)3 * MROW * DP * 2);
  float*          S   = (float*)take((size_t)3 * BB * LL * LL * 4);
  unsigned short* X3  = (unsigned short*)take((size_t)3 * BB * LL * ABH * 2);
  unsigned short* X1c = (unsigned short*)take((size_t)MROW * RIN * 2);
  unsigned short* Wih = (unsigned short*)take((size_t)2 * G4 * RIN * 2);
  unsigned short* Whh = (unsigned short*)take((size_t)2 * G4 * HH * 2);
  float*          bia = (float*)take((size_t)2 * G4 * 4);
  float*          Z   = (float*)take((size_t)2 * LL * BB * G4 * 4);

  const int EB = 256;
  // 1) pack / convert
  k_pack_att<<<8192, EB, 0, stream>>>(x1a, x1w, x1a0, x1a1);
  k_pack_att<<<8192, EB, 0, stream>>>(x2a, x2w, x2a0, x2a1);
  for (int h = 0; h < 3; ++h)
    k_pack_U<<<1344, EB, 0, stream>>>(Ubf + (size_t)h * DP * ATTP, Us[h]);
  const float* x3src[3] = {x2a0, x2a1, x2a2};
  for (int h = 0; h < 3; ++h)
    k_f2bf<<<8192, EB, 0, stream>>>(X3 + (size_t)h * BB * LL * ABH, x3src[h], (size_t)BB * LL * ABH);
  k_pack_x1cat<<<8192, EB, 0, stream>>>(X1c, x1a0, x1a1);
  k_f2bf<<<8192, EB, 0, stream>>>(Wih, Wih_f, (size_t)G4 * RIN);
  k_f2bf<<<8192, EB, 0, stream>>>(Wih + (size_t)G4 * RIN, Wih_b, (size_t)G4 * RIN);
  k_f2bf<<<1024, EB, 0, stream>>>(Whh, Whh_f, (size_t)G4 * HH);
  k_f2bf<<<1024, EB, 0, stream>>>(Whh + (size_t)G4 * HH, Whh_b, (size_t)G4 * HH);
  k_bias<<<4, EB, 0, stream>>>(bia, bih_f, bhh_f);
  k_bias<<<4, EB, 0, stream>>>(bia + G4, bih_b, bhh_b);

  // 2) projections (WMMA, 16x64 per wave)
  for (int h = 0; h < 3; ++h) {
    k_proj<<<dim3(DP / 64, MROW / 16, 1), 32, 0, stream>>>(
        x1a, Ubf + (size_t)h * DP * ATTP, ds[h], R1 + (size_t)h * MROW * DP, 0);
    k_proj<<<dim3(DP / 64, MROW / 16, 1), 32, 0, stream>>>(
        x2a, Ubf + (size_t)h * DP * ATTP, ds[h], R2 + (size_t)h * MROW * DP, 1);
  }
  // 3) scores + mask (WMMA)
  k_scores<<<dim3(LL / 64, LL / 16, 3 * BB), 32, 0, stream>>>(R1, R2, mask, S);
  // 4) softmax -> bf16 alpha in place
  k_softmax<<<(3 * BB * LL) / 8, EB, 0, stream>>>(S);
  // 5) alpha @ x3 (WMMA) into x1cat
  for (int h = 0; h < 3; ++h)
    k_attnv<<<dim3(ABH / 64, LL / 16, BB), 32, 0, stream>>>((const unsigned short*)S, X3, X1c, h);
  // 6) LSTM input projection (WMMA + async LDS double buffering)
  k_zproj<<<dim3(G4 / 64, (LL * BB / 16), 2), 32, 0, stream>>>(X1c, Wih, bia, Z);
  // 7) recurrence (persistent, WMMA from LDS)
  k_lstm<<<2, 1024, 0, stream>>>(Z, Whh, (float*)d_out);
}
